// CausalSelfAttention_23106924053318
// MI455X (gfx1250) — matrix operand order repaired and checked by reading, not compiled, not run on hardware
//
#include <hip/hip_runtime.h>

typedef __attribute__((ext_vector_type(8)))  _Float16 v8h;
typedef __attribute__((ext_vector_type(16))) _Float16 v16h;
typedef __attribute__((ext_vector_type(8)))  float    v8f;

#define WMMA_F16(a, b, c) \
  __builtin_amdgcn_wmma_f32_16x16x32_f16(false, (a), false, (b), (short)0, (c), false, false)

constexpr int Bsz = 4, Sq = 2048, Dm = 1024, Hh = 16, HD = 64;
constexpr int Mrows = Bsz * Sq;   // 8192
constexpr int N3 = 3 * Dm;        // 3072

// ---- WMMA fragment loaders (CDNA5 ISA 7.12.2 layouts, wave32) ----
// A-matrix 16x32 f16: lane r=lane&15 holds row r; K pattern: kb=(lane>=16?8:0),
// halves [kb..kb+7] then [kb+16..kb+23]. Caller passes rowbase + kchunk + kb.
__device__ __forceinline__ v16h ldA(const _Float16* p) {
  v16h a;
  *(v8h*)&a       = *(const v8h*)p;
  *((v8h*)&a + 1) = *(const v8h*)(p + 16);
  return a;
}
// B-matrix 32x16 f16: lane holds column n=lane&15 of B (= row n of B^T);
// K pattern same as A: kb=(lane>=16?8:0), halves [kb..kb+7] then [kb+16..kb+23].
// Caller passes colbase + kchunk + kb.
__device__ __forceinline__ v16h ldB(const _Float16* p) {
  v16h a;
  *(v8h*)&a       = *(const v8h*)p;
  *((v8h*)&a + 1) = *(const v8h*)(p + 16);
  return a;
}

__global__ void cvt_f16(const float* __restrict__ in, _Float16* __restrict__ out, int n) {
  int i = blockIdx.x * blockDim.x + threadIdx.x;
  if (i < n) out[i] = (_Float16)in[i];
}

// qkv = x @ w_qkv^T + b_qkv ; scatter into Q[b,h,s,d], K[b,h,s,d], VT[b,h,d,s] (f16)
// Wave tile: 32(M) x 64(N); block = 8 waves -> 32 x 512 block tile.
__global__ __launch_bounds__(256) void qkv_gemm(
    const _Float16* __restrict__ X, const _Float16* __restrict__ W,
    const float* __restrict__ bias,
    _Float16* __restrict__ Qh, _Float16* __restrict__ Kh, _Float16* __restrict__ VTh)
{
  const int lane = threadIdx.x & 31;
  const int wave = threadIdx.x >> 5;
  const int m0 = blockIdx.y * 32;
  const int n0 = blockIdx.x * 512 + wave * 64;
  const int r    = lane & 15;
  const int kb   = (lane >> 4) << 3;
  const int kb16 = (lane >> 4) << 4;
  const int jb   = (lane >> 4) << 3;   // C-layout row base

  v8f c[2][4];
  #pragma unroll
  for (int i = 0; i < 2; ++i)
    #pragma unroll
    for (int j = 0; j < 4; ++j) c[i][j] = (v8f){};

  const _Float16* Ar0 = X + (size_t)(m0 + r) * Dm;
  const _Float16* Ar1 = X + (size_t)(m0 + 16 + r) * Dm;
  const _Float16* Br[4] = {
    W + (size_t)(n0 + 0 * 16 + r) * Dm,
    W + (size_t)(n0 + 1 * 16 + r) * Dm,
    W + (size_t)(n0 + 2 * 16 + r) * Dm,
    W + (size_t)(n0 + 3 * 16 + r) * Dm };

  for (int kk = 0; kk < Dm; kk += 32) {
    v16h a0 = ldA(Ar0 + kk + kb);
    v16h a1 = ldA(Ar1 + kk + kb);
    #pragma unroll
    for (int j = 0; j < 4; ++j) {
      v16h b = ldB(Br[j] + kk + kb);
      c[0][j] = WMMA_F16(a0, b, c[0][j]);
      c[1][j] = WMMA_F16(a1, b, c[1][j]);
    }
  }
  #pragma unroll
  for (int i = 0; i < 2; ++i) {
    #pragma unroll
    for (int j = 0; j < 4; ++j) {
      #pragma unroll
      for (int t = 0; t < 8; ++t) {
        int m = m0 + i * 16 + jb + t;
        int n = n0 + j * 16 + r;
        float v = c[i][j][t] + bias[n];
        int b = m >> 11, s = m & (Sq - 1);
        _Float16 hv = (_Float16)v;
        if (n < Dm) {
          int h = n >> 6, d = n & 63;
          Qh[(((size_t)(b * Hh + h) * Sq) + s) * HD + d] = hv;
        } else if (n < 2 * Dm) {
          int nn = n - Dm; int h = nn >> 6, d = nn & 63;
          Kh[(((size_t)(b * Hh + h) * Sq) + s) * HD + d] = hv;
        } else {
          int nn = n - 2 * Dm; int h = nn >> 6, d = nn & 63;
          VTh[((size_t)(b * Hh + h) * HD + d) * Sq + s] = hv;
        }
      }
    }
  }
}

// Flash attention: one wave per 16-query tile, online softmax over 32-key blocks.
__global__ __launch_bounds__(256) void attn(
    const _Float16* __restrict__ Qh, const _Float16* __restrict__ Kh,
    const _Float16* __restrict__ VTh, _Float16* __restrict__ Yh)
{
  __shared__ _Float16 pbuf[8][16 * 40];   // per-wave 16x32 P tile, stride 40 halves
  const int lane = threadIdx.x & 31;
  const int wave = threadIdx.x >> 5;
  const int qt = blockIdx.x * 8 + wave;   // 0..127
  const int bh = blockIdx.y;              // 0..63
  const int q0 = qt * 16;
  const int r    = lane & 15;
  const int kb   = (lane >> 4) << 3;
  const int kb16 = (lane >> 4) << 4;
  const int jb   = kb;

  const _Float16* Qb = Qh  + (size_t)bh * Sq * HD;
  const _Float16* Kb = Kh  + (size_t)bh * Sq * HD;
  const _Float16* Vb = VTh + (size_t)bh * HD * Sq;

  v16h aq0 = ldA(Qb + (size_t)(q0 + r) * HD + kb);
  v16h aq1 = ldA(Qb + (size_t)(q0 + r) * HD + 32 + kb);

  v8f o0 = {}, o1 = {}, o2 = {}, o3 = {};
  float mrow[8], lrow[8];
  #pragma unroll
  for (int j = 0; j < 8; ++j) { mrow[j] = -3.0e38f; lrow[j] = 0.0f; }

  _Float16* P = &pbuf[wave][0];
  const int np = (qt >> 1) + 1;
  for (int kp = 0; kp < np; ++kp) {
    const int key0 = kp * 32;
    v16h bk0 = ldB(Kb + (size_t)(key0 + r) * HD + kb);
    v16h bk1 = ldB(Kb + (size_t)(key0 + r) * HD + 32 + kb);
    v16h bk2 = ldB(Kb + (size_t)(key0 + 16 + r) * HD + kb);
    v16h bk3 = ldB(Kb + (size_t)(key0 + 16 + r) * HD + 32 + kb);
    v8f s0 = {}, s1 = {};
    s0 = WMMA_F16(aq0, bk0, s0);
    s0 = WMMA_F16(aq1, bk1, s0);
    s1 = WMMA_F16(aq0, bk2, s1);
    s1 = WMMA_F16(aq1, bk3, s1);

    const bool last = (kp == np - 1);
    float x0[8], x1[8], mnew[8], alpha[8];
    #pragma unroll
    for (int j = 0; j < 8; ++j) {
      float t0 = s0[j] * 0.125f;   // 1/sqrt(64)
      float t1 = s1[j] * 0.125f;
      if (last) {
        int qrow = q0 + jb + j;
        t0 = (key0 + r)      > qrow ? -3.0e38f : t0;
        t1 = (key0 + 16 + r) > qrow ? -3.0e38f : t1;
      }
      x0[j] = t0; x1[j] = t1;
      float v = fmaxf(t0, t1);      // reduce across the 16 lanes of this row group
      v = fmaxf(v, __shfl_xor(v, 1, 32));
      v = fmaxf(v, __shfl_xor(v, 2, 32));
      v = fmaxf(v, __shfl_xor(v, 4, 32));
      v = fmaxf(v, __shfl_xor(v, 8, 32));
      mnew[j] = fmaxf(mrow[j], v);
      alpha[j] = __expf(mrow[j] - mnew[j]);
      mrow[j] = mnew[j];
    }
    #pragma unroll
    for (int j = 0; j < 8; ++j) {
      float e0 = __expf(x0[j] - mnew[j]);
      float e1 = __expf(x1[j] - mnew[j]);
      float rs = e0 + e1;
      rs += __shfl_xor(rs, 1, 32);
      rs += __shfl_xor(rs, 2, 32);
      rs += __shfl_xor(rs, 4, 32);
      rs += __shfl_xor(rs, 8, 32);
      lrow[j] = lrow[j] * alpha[j] + rs;
      P[(jb + j) * 40 + r]      = (_Float16)e0;
      P[(jb + j) * 40 + 16 + r] = (_Float16)e1;
      o0[j] *= alpha[j]; o1[j] *= alpha[j]; o2[j] *= alpha[j]; o3[j] *= alpha[j];
    }
    v16h ap  = ldA(P + r * 40 + kb);                       // LDS round-trip: C-layout -> A-fragment
    v16h bv0 = ldB(Vb + (size_t)(0 * 16 + r) * Sq + key0 + kb);
    v16h bv1 = ldB(Vb + (size_t)(1 * 16 + r) * Sq + key0 + kb);
    v16h bv2 = ldB(Vb + (size_t)(2 * 16 + r) * Sq + key0 + kb);
    v16h bv3 = ldB(Vb + (size_t)(3 * 16 + r) * Sq + key0 + kb);
    o0 = WMMA_F16(ap, bv0, o0);
    o1 = WMMA_F16(ap, bv1, o1);
    o2 = WMMA_F16(ap, bv2, o2);
    o3 = WMMA_F16(ap, bv3, o3);
  }

  const int b = bh >> 4, h = bh & 15;
  #pragma unroll
  for (int j = 0; j < 8; ++j) {
    float inv = 1.0f / lrow[j];
    int s = q0 + jb + j;
    _Float16* yrow = Yh + (size_t)(b * Sq + s) * Dm + h * HD;
    yrow[0 * 16 + r] = (_Float16)(o0[j] * inv);
    yrow[1 * 16 + r] = (_Float16)(o1[j] * inv);
    yrow[2 * 16 + r] = (_Float16)(o2[j] * inv);
    yrow[3 * 16 + r] = (_Float16)(o3[j] * inv);
  }
}

// out = y @ w_out^T + b_out (f32 output). Wave tile: 32(M) x 64(N).
__global__ __launch_bounds__(256) void out_gemm(
    const _Float16* __restrict__ Y, const _Float16* __restrict__ W,
    const float* __restrict__ bias, float* __restrict__ out)
{
  const int lane = threadIdx.x & 31;
  const int wave = threadIdx.x >> 5;
  const int m0 = blockIdx.y * 32;
  const int n0 = blockIdx.x * 512 + wave * 64;
  const int r    = lane & 15;
  const int kb   = (lane >> 4) << 3;
  const int kb16 = (lane >> 4) << 4;
  const int jb   = kb;

  v8f c[2][4];
  #pragma unroll
  for (int i = 0; i < 2; ++i)
    #pragma unroll
    for (int j = 0; j < 4; ++j) c[i][j] = (v8f){};

  const _Float16* Ar0 = Y + (size_t)(m0 + r) * Dm;
  const _Float16* Ar1 = Y + (size_t)(m0 + 16 + r) * Dm;
  const _Float16* Br[4] = {
    W + (size_t)(n0 + 0 * 16 + r) * Dm,
    W + (size_t)(n0 + 1 * 16 + r) * Dm,
    W + (size_t)(n0 + 2 * 16 + r) * Dm,
    W + (size_t)(n0 + 3 * 16 + r) * Dm };

  for (int kk = 0; kk < Dm; kk += 32) {
    v16h a0 = ldA(Ar0 + kk + kb);
    v16h a1 = ldA(Ar1 + kk + kb);
    #pragma unroll
    for (int j = 0; j < 4; ++j) {
      v16h b = ldB(Br[j] + kk + kb);
      c[0][j] = WMMA_F16(a0, b, c[0][j]);
      c[1][j] = WMMA_F16(a1, b, c[1][j]);
    }
  }
  #pragma unroll
  for (int i = 0; i < 2; ++i) {
    #pragma unroll
    for (int j = 0; j < 4; ++j) {
      #pragma unroll
      for (int t = 0; t < 8; ++t) {
        int m = m0 + i * 16 + jb + t;
        int n = n0 + j * 16 + r;
        out[(size_t)m * Dm + n] = c[i][j][t] + bias[n];
      }
    }
  }
}

extern "C" void kernel_launch(void* const* d_in, const int* in_sizes, int n_in,
                              void* d_out, int out_size, void* d_ws, size_t ws_size,
                              hipStream_t stream) {
  (void)in_sizes; (void)n_in; (void)out_size; (void)ws_size;
  const float* x     = (const float*)d_in[0];
  const float* w_qkv = (const float*)d_in[1];
  const float* b_qkv = (const float*)d_in[2];
  const float* w_out = (const float*)d_in[3];
  const float* b_out = (const float*)d_in[4];
  float* out = (float*)d_out;

  char* ws = (char*)d_ws;
  size_t off = 0;
  auto take = [&](size_t elems) {
    _Float16* p = (_Float16*)(ws + off);
    off += ((elems * sizeof(_Float16) + 255) & ~(size_t)255);
    return p;
  };
  _Float16* xh  = take((size_t)Mrows * Dm);  // x in f16
  _Float16* wqh = take((size_t)N3 * Dm);     // w_qkv f16
  _Float16* woh = take((size_t)Dm * Dm);     // w_out f16
  _Float16* qhd = take((size_t)Mrows * Dm);  // Q [b,h,s,d]
  _Float16* khd = take((size_t)Mrows * Dm);  // K [b,h,s,d]
  _Float16* vth = take((size_t)Mrows * Dm);  // V^T [b,h,d,s]
  _Float16* yh  = take((size_t)Mrows * Dm);  // attention output [b,s,D]

  const int nx  = Mrows * Dm;
  const int nwq = N3 * Dm;
  const int nwo = Dm * Dm;
  cvt_f16<<<dim3((nx  + 255) / 256), dim3(256), 0, stream>>>(x,     xh,  nx);
  cvt_f16<<<dim3((nwq + 255) / 256), dim3(256), 0, stream>>>(w_qkv, wqh, nwq);
  cvt_f16<<<dim3((nwo + 255) / 256), dim3(256), 0, stream>>>(w_out, woh, nwo);

  qkv_gemm<<<dim3(N3 / 512, Mrows / 32), dim3(256), 0, stream>>>(xh, wqh, b_qkv, qhd, khd, vth);
  attn<<<dim3((Sq / 16) / 8, Bsz * Hh), dim3(256), 0, stream>>>(qhd, khd, vth, yh);
  out_gemm<<<dim3(Dm / 512, Mrows / 32), dim3(256), 0, stream>>>(yh, woh, b_out, out);
}
